// GlobalAggregator_79173427134504
// MI455X (gfx1250) — matrix operand order repaired and checked by reading, not compiled, not run on hardware
//
#include <hip/hip_runtime.h>

typedef __attribute__((ext_vector_type(16))) _Float16 v16h;
typedef __attribute__((ext_vector_type(4)))  _Float16 h4;
typedef __attribute__((ext_vector_type(8)))  float    v8f;
typedef __attribute__((ext_vector_type(4)))  unsigned int u32x4;
typedef __attribute__((ext_vector_type(4)))  int i32x4;
typedef __attribute__((ext_vector_type(8)))  int i32x8;

#define DIM 128
#define KNB 16
#define WAVES_PER_BLOCK 8
#define NB_PITCH 132  // padded neighbor row pitch (floats) -> LDS bank-conflict free

// generic -> concrete address-space casts (through integer to sidestep constness).
// The async-to-LDS builtin takes (int4 AS1*, int4 AS3*, imm, imm) per the probe.
#define AS_GLOBAL_I4(p) ((__attribute__((address_space(1))) i32x4*)(unsigned long long)(size_t)(const void*)(p))
#define AS_LDS_I4(p)    ((__attribute__((address_space(3))) i32x4*)(unsigned)(size_t)(const void*)(p))

// ---------------------------------------------------------------------------
// Prep: swizzle w_1[0:128,0:128] (fp32, row = inner d, col = out feature e)
// into WMMA B-fragment f16 layout: index = ((t*4+c)*32 + lane)*16 + e,
// K = c*32 + (lane>>4)*16 + e, N = t*16 + (lane&15). 16384 halves = 32 KB.
// ---------------------------------------------------------------------------
__global__ void prep_w1_swizzle(const float* __restrict__ w1,
                                _Float16* __restrict__ w1b) {
    int idx  = blockIdx.x * blockDim.x + threadIdx.x;   // 0..16383
    int e    = idx & 15;
    int lane = (idx >> 4) & 31;
    int c    = (idx >> 9) & 3;
    int t    = idx >> 11;
    int kk = c * 32 + (lane >> 4) * 16 + e;
    int n  = t * 16 + (lane & 15);
    w1b[idx] = (_Float16)w1[kk * DIM + n];
}

// ---------------------------------------------------------------------------
// Main: one wave32 per (b,n) pair, 8 waves / block.
// ---------------------------------------------------------------------------
__global__ void __launch_bounds__(256) gat_aggregate(
    const float*     __restrict__ neighbor,  // [BN,16,128]
    const float*     __restrict__ nweight,   // [BN,16]
    const float*     __restrict__ extra,     // [BN,128]
    const float*     __restrict__ w1_last,   // [128] = w_1[-1]
    const float*     __restrict__ w2,        // [128]
    const _Float16*  __restrict__ w1b,       // [16384] swizzled B fragments
    float*           __restrict__ out)       // [BN,128]
{
    extern __shared__ char smem[];
    _Float16* sW1 = (_Float16*)smem;                                   // 32 KB
    float*    sNb = (float*)(smem + 32768);                            // 8*16*132 f32
    _Float16* sA  = (_Float16*)(smem + 32768 +
                                WAVES_PER_BLOCK * KNB * NB_PITCH * 4); // 8*2048 f16

    const int tid  = threadIdx.x;
    const int wave = tid >> 5;
    const int lane = tid & 31;
    const int half = lane >> 4;
    const int lr   = lane & 15;
    const int bn   = blockIdx.x * WAVES_PER_BLOCK + wave;

    const float* nb   = neighbor + (size_t)bn * (KNB * DIM);
    const float* ex   = extra    + (size_t)bn * DIM;
    const float* wk   = nweight  + (size_t)bn * KNB;
    float*       outp = out      + (size_t)bn * DIM;

    // ---------------- w1 fragments: global -> LDS (TDM) --------------------
#if __has_builtin(__builtin_amdgcn_tensor_load_to_lds)
    if (wave == 0) {
        // TDM: 1-D tile of 8192 dwords (32 KB), contiguous.
        u32x4 g0 = {}; i32x8 g1 = {}; i32x4 g2 = {}, g3 = {};
        unsigned long long ga = (unsigned long long)(size_t)(const void*)w1b;
        g0[0] = 1u;                                              // count = 1 (valid)
        g0[1] = (unsigned)(size_t)(void*)sW1;                    // lds_addr
        g0[2] = (unsigned)ga;                                    // global_addr lo
        g0[3] = ((unsigned)(ga >> 32) & 0x01FFFFFFu) | (2u << 30); // addr hi | type=2
        g1[0] = 2 << 16;                                         // data_size = 4 B
        g1[1] = (int)(8192u << 16);                              // tensor_dim0 = 8192
        g1[2] = (int)(1u << 16);                                 // tensor_dim1 = 1
        g1[3] = (int)(8192u << 16);                              // tile_dim0 = 8192
        g1[5] = 8192;                                            // tensor_dim0_stride
#if defined(__clang_major__) && __clang_major__ >= 23
        i32x8 g4 = {};
        __builtin_amdgcn_tensor_load_to_lds(g0, g1, g2, g3, g4, 0);
#else
        __builtin_amdgcn_tensor_load_to_lds(g0, g1, g2, g3, 0);
#endif
        __builtin_amdgcn_s_wait_tensorcnt(0);
        asm volatile("" ::: "memory");
    }
#else
    {   // fallback: cooperative float4 copy of 32 KB
        const float4* src = (const float4*)w1b;
        float4*       dst = (float4*)sW1;
        #pragma unroll
        for (int i = 0; i < 8; ++i) dst[tid + 256 * i] = src[tid + 256 * i];
    }
#endif

    // ---------------- neighbor tile: global -> LDS (async) ----------------
    float* myNb = sNb + wave * (KNB * NB_PITCH);
    const int col = lane * 4;                  // this lane's 4 features

#if __has_builtin(__builtin_amdgcn_global_load_async_to_lds_b128)
    #pragma unroll
    for (int j = 0; j < KNB; ++j) {            // row j, 16 B per lane
        __builtin_amdgcn_global_load_async_to_lds_b128(
            AS_GLOBAL_I4(nb + j * DIM + col),
            AS_LDS_I4(myNb + j * NB_PITCH + col), 0, 0);
    }
    float4 ex4 = ((const float4*)ex)[lane];    // overlap with async copies
#if __has_builtin(__builtin_amdgcn_s_wait_asynccnt)
    __builtin_amdgcn_s_wait_asynccnt(0);
#else
    asm volatile("s_wait_asynccnt 0x0" ::: "memory");
#endif
    asm volatile("" ::: "memory");
#else
    float4 ex4 = ((const float4*)ex)[lane];
    #pragma unroll
    for (int j = 0; j < KNB; ++j) {
        float4 v = ((const float4*)nb)[j * 32 + lane];
        *(float4*)&myNb[j * NB_PITCH + col] = v;
    }
#endif

    // ---------------- build gated f16 A-fragments in LDS -------------------
    // col -> (chunk c, K-half hd, element e0); a float4 = 4 consecutive halves.
    _Float16* aW = sA + wave * (4 * 32 * 16);
    {
        const int c_ = col >> 5;
        const int dd = col & 31;
        const int hd = dd >> 4;
        const int e0 = dd & 15;
        _Float16* aBase = aW + (c_ * 32 + hd * 16) * 16 + e0;
        #pragma unroll
        for (int j = 0; j < KNB; ++j) {        // row k = j -> lane slot hd*16+j
            float4 v = *(const float4*)&myNb[j * NB_PITCH + col];
            h4 g;
            g[0] = (_Float16)(v.x * ex4.x);
            g[1] = (_Float16)(v.y * ex4.y);
            g[2] = (_Float16)(v.z * ex4.z);
            g[3] = (_Float16)(v.w * ex4.w);
            *(h4*)(aBase + j * 16) = g;        // ds_store_b64
        }
    }
    __syncthreads();

    // ---------------- preload fragments and epilogue constants -------------
    v16h a[4];
    const v16h* aFrag = (const v16h*)aW;
    #pragma unroll
    for (int c = 0; c < 4; ++c) a[c] = aFrag[c * 32 + lane];   // 2x ds_load_b128

    float wgt[8];
    #pragma unroll
    for (int i = 0; i < 8; ++i) wgt[i] = wk[half * 8 + i];

    float biasv[8], w2v[8];
    #pragma unroll
    for (int t = 0; t < 8; ++t) {
        int n = t * 16 + lr;
        biasv[t] = w1_last[n];
        w2v[t]   = w2[n];
    }

    // ---------------- 8 tiles x 4 chunks of v_wmma_f32_16x16x32_f16 --------
    float sp[8];
    #pragma unroll
    for (int i = 0; i < 8; ++i) sp[i] = 0.0f;

    const v16h* bfrag = (const v16h*)sW1;
    #pragma unroll
    for (int t = 0; t < 8; ++t) {
        v16h b0 = bfrag[(t * 4 + 0) * 32 + lane];
        v16h b1 = bfrag[(t * 4 + 1) * 32 + lane];
        v16h b2 = bfrag[(t * 4 + 2) * 32 + lane];
        v16h b3 = bfrag[(t * 4 + 3) * 32 + lane];
        v8f acc = {};
        acc = __builtin_amdgcn_wmma_f32_16x16x32_f16(false, a[0], false, b0, (short)0, acc, false, false);
        acc = __builtin_amdgcn_wmma_f32_16x16x32_f16(false, a[1], false, b1, (short)0, acc, false, false);
        acc = __builtin_amdgcn_wmma_f32_16x16x32_f16(false, a[2], false, b2, (short)0, acc, false, false);
        acc = __builtin_amdgcn_wmma_f32_16x16x32_f16(false, a[3], false, b3, (short)0, acc, false, false);
        // fused epilogue: bias, leaky_relu(0.2), contract with w2
        #pragma unroll
        for (int i = 0; i < 8; ++i) {
            float v = acc[i] + wgt[i] * biasv[t];   // D[i] -> (k=half*8+i, n=t*16+lr)
            v = (v > 0.0f) ? v : 0.2f * v;
            sp[i] += v * w2v[t];
        }
    }

    // ---------------- score reduction + softmax (wave32 shuffles) ----------
    #pragma unroll
    for (int m = 1; m < 16; m <<= 1) {
        #pragma unroll
        for (int i = 0; i < 8; ++i) sp[i] += __shfl_xor(sp[i], m, 32);
    }
    float sc[16];
    #pragma unroll
    for (int i = 0; i < 8; ++i) {
        float other = __shfl_xor(sp[i], 16, 32);
        sc[half * 8 + i]       = sp[i];
        sc[(1 - half) * 8 + i] = other;
    }

    float mx = sc[0];
    #pragma unroll
    for (int k = 1; k < 16; ++k) mx = fmaxf(mx, sc[k]);
    float sum = 0.0f;
    float sw[16];
    #pragma unroll
    for (int k = 0; k < 16; ++k) { sw[k] = __expf(sc[k] - mx); sum += sw[k]; }
    float inv = 1.0f / sum;

    // ---------------- out[d] = sum_k softmax[k] * neighbor[k][d] -----------
    float4 o = {0.f, 0.f, 0.f, 0.f};
    #pragma unroll
    for (int k = 0; k < 16; ++k) {
        float4 v = *(const float4*)&myNb[k * NB_PITCH + col];
        float  s = sw[k] * inv;
        o.x += s * v.x; o.y += s * v.y; o.z += s * v.z; o.w += s * v.w;
    }
    ((float4*)outp)[lane] = o;
}

// ---------------------------------------------------------------------------
// Launch.  inputs: 0 self_vectors, 1 neighbor_vector, 2 batch_size, 3 masks,
//                  4 neighbor_weight, 5 extra_vector, 6 w_1, 7 w_2
// ---------------------------------------------------------------------------
extern "C" void kernel_launch(void* const* d_in, const int* in_sizes, int n_in,
                              void* d_out, int out_size, void* d_ws, size_t ws_size,
                              hipStream_t stream) {
    const float* neighbor = (const float*)d_in[1];
    const float* nweight  = (const float*)d_in[4];
    const float* extra    = (const float*)d_in[5];
    const float* w1       = (const float*)d_in[6];   // [129,128]
    const float* w2       = (const float*)d_in[7];   // [128,1]
    float*       outp     = (float*)d_out;
    _Float16*    w1b      = (_Float16*)d_ws;         // 32 KB scratch

    prep_w1_swizzle<<<64, 256, 0, stream>>>(w1, w1b);

    const int BN     = 256 * 256;
    const int blocks = BN / WAVES_PER_BLOCK;                         // 8192
    const size_t lds = 32768                                          // w1 frags
                     + WAVES_PER_BLOCK * KNB * NB_PITCH * sizeof(float) // nb tiles
                     + WAVES_PER_BLOCK * 4 * 32 * 16 * sizeof(_Float16); // A frags
    gat_aggregate<<<blocks, 256, lds, stream>>>(
        neighbor, nweight, extra,
        w1 + DIM * DIM,   // bias row w_1[-1]
        w2, w1b, outp);
}
